// SquareLoss_12163347382641
// MI455X (gfx1250) — matrix operand order repaired and checked by reading, not compiled, not run on hardware
//
#include <hip/hip_runtime.h>
#include <hip/hip_bf16.h>

typedef __attribute__((ext_vector_type(16))) _Float16 v16h;
typedef __attribute__((ext_vector_type(8)))  float    v8f;

#define NPIX   65536   // H*W
#define NCH    64
#define NBATCH 8
#define NCLS   4
#define EPSN   1e-12f
#define TILES_PER_PAIR 32          // 32 blocks of 2048 pixels each
#define PIX_PER_BLOCK  2048

// workspace layout (floats):
//  [0,    2048)  y vectors      (k*8+b)*64 + c
//  [2048, 2080)  err flags      (k*8+b)
//  [2080, 3104)  block partials (k*8+b)*32 + tile

__device__ __forceinline__ float hash01(unsigned x) {
    x ^= x >> 16; x *= 0x7feb352dU;
    x ^= x >> 15; x *= 0x846ca68bU;
    x ^= x >> 16;
    return (float)(x >> 8) * (1.0f / 16777216.0f);
}

// ---------------- Kernel 1: per-(k,b) rank-sample + normalize y ----------------
__global__ __launch_bounds__(256) void sample_y_kernel(
    const float* __restrict__ f1_0, const float* __restrict__ f1_1,
    const float* __restrict__ f1_2, const float* __restrict__ f1_3,
    const int*   __restrict__ t1,
    float* __restrict__ y_ws, float* __restrict__ err_ws)
{
    const int b = blockIdx.x;            // 0..7
    const int k = blockIdx.y;            // 0..3
    const float* f1 = (k == 0) ? f1_0 : (k == 1) ? f1_1 : (k == 2) ? f1_2 : f1_3;
    const int* t1b = t1 + (size_t)b * NPIX;
    const int tid = threadIdx.x;
    const int chunk = NPIX / 256;        // 256 pixels per thread

    __shared__ int cnts[256];
    __shared__ int pref[257];
    __shared__ int s_idx;
    __shared__ float vsh[NCH];
    __shared__ float s_sumsq;

    int start = tid * chunk;
    int c = 0;
    for (int i = 0; i < chunk; ++i) c += (t1b[start + i] == k) ? 1 : 0;
    cnts[tid] = c;
    __syncthreads();

    if (tid == 0) {                       // cheap serial exclusive scan (256 elems)
        int s = 0;
        for (int i = 0; i < 256; ++i) { pref[i] = s; s += cnts[i]; }
        pref[256] = s;
        s_idx = 0;
    }
    __syncthreads();

    const int total = pref[256];
    const int err   = (total == 0) ? 1 : 0;
    const float u   = hash01(42u * 4096u + (unsigned)k * 256u + (unsigned)b + 1u);
    int r = (int)floorf(u * (float)max(total, 1));
    r = min(r, max(total - 1, 0));

    if (!err && r >= pref[tid] && r < pref[tid] + cnts[tid]) {
        int need = r - pref[tid];
        for (int i = 0; i < chunk; ++i) {
            if (t1b[start + i] == k) {
                if (need == 0) { s_idx = start + i; break; }
                --need;
            }
        }
    }
    __syncthreads();
    const int idx = s_idx;

    if (tid < NCH) {
        float v = err ? 0.0f
                      : f1[((size_t)b * NCH + tid) * NPIX + idx];
        vsh[tid] = v;
    }
    __syncthreads();
    if (tid == 0) {
        float s = 0.0f;
        for (int i = 0; i < NCH; ++i) s += vsh[i] * vsh[i];
        s_sumsq = s;
    }
    __syncthreads();
    if (tid < NCH) {
        float denom = fmaxf(sqrtf(s_sumsq), EPSN);
        y_ws[((size_t)k * NBATCH + b) * NCH + tid] = vsh[tid] / denom;
    }
    if (tid == 0) err_ws[k * NBATCH + b] = (float)err;
}

// ---------------- Kernel 2: fused normalize + GEMV(WMMA) + loss partials -------
__global__ __launch_bounds__(256) void inner_loss_kernel(
    const float* __restrict__ f2_0, const float* __restrict__ f2_1,
    const float* __restrict__ f2_2, const float* __restrict__ f2_3,
    const int*   __restrict__ t2,
    const float* __restrict__ y_ws, float* __restrict__ partial_ws)
{
    const int tile = blockIdx.x;          // 0..31
    const int b    = blockIdx.y;          // 0..7
    const int k    = blockIdx.z;          // 0..3
    const float* f2 = (k == 0) ? f2_0 : (k == 1) ? f2_1 : (k == 2) ? f2_2 : f2_3;
    const float* fb  = f2 + (size_t)b * NCH * NPIX;
    const int*   t2b = t2 + (size_t)b * NPIX;

    const int tid  = threadIdx.x;
    const int lane = tid & 31;
    const int wave = tid >> 5;
    const int nIdx = lane & 15;           // pixel within 16-wide tile
    const bool hi  = lane >= 16;          // lanes L and L^16 share a pixel,
    const int  kb  = hi ? 16 : 0;         // complementary 32-channel halves

    __shared__ float ysh[NCH];
    if (tid < NCH) ysh[tid] = y_ws[((size_t)k * NBATCH + b) * NCH + tid];
    __syncthreads();

    // A fragment: y broadcast to all 16 rows.
    // 16-bit A 16x32 layout (contiguous, matches B): lanes 0-15 hold K 0..15, lanes 16-31 K 16..31
    v16h a_lo{}, a_hi{};
#pragma unroll
    for (int e = 0; e < 16; ++e) {
        int K = e + (hi ? 16 : 0);
        a_lo[e] = (_Float16)ysh[K];
        a_hi[e] = (_Float16)ysh[32 + K];
    }

    float acc = 0.0f;
    const int pBlock = tile * PIX_PER_BLOCK;

    for (int it = 0; it < PIX_PER_BLOCK / 128; ++it) {   // 16 iterations
        const int p = pBlock + it * 128 + wave * 16 + nIdx;

        // Load this lane's 32 channels of pixel p (f32), track sum of squares.
        float vals[32];
        float ss = 0.0f;
#pragma unroll
        for (int j = 0; j < 16; ++j) {
            float v = fb[(size_t)(kb + j) * NPIX + p];
            vals[j] = v; ss += v * v;
        }
#pragma unroll
        for (int j = 0; j < 16; ++j) {
            float v = fb[(size_t)(32 + kb + j) * NPIX + p];
            vals[16 + j] = v; ss += v * v;
        }

        // B fragments: 32x16 f16. Lanes 0-15: K=0..15; lanes 16-31: K=16..31.
        v16h b_lo{}, b_hi{};
#pragma unroll
        for (int j = 0; j < 16; ++j) {
            b_lo[j] = (_Float16)vals[j];
            b_hi[j] = (_Float16)vals[16 + j];
        }

        v8f cacc = {};
        cacc = __builtin_amdgcn_wmma_f32_16x16x32_f16(
            false, a_lo, false, b_lo, (short)0, cacc, false, false);
        cacc = __builtin_amdgcn_wmma_f32_16x16x32_f16(
            false, a_hi, false, b_hi, (short)0, cacc, false, false);
        float d = cacc[0];   // M=0 row (lanes 0-15) / M=8 row (lanes 16-31): identical

        // full per-pixel squared norm: combine complementary channel halves.
        // All 32 lanes must execute the shuffle (hi lanes supply their half).
        float sumsq = ss + __shfl_xor(ss, 16, 32);

        if (!hi) {
            // Fast hw approximations: v_sqrt_f32 (~1 ulp) + v_rcp_f32 (~2 ulp).
            // Dot is already f16-precision; this is not the accuracy limiter.
            float denom = fmaxf(__builtin_amdgcn_sqrtf(sumsq), EPSN);
            float inner = d * __builtin_amdgcn_rcpf(denom);
            float mapk  = (t2b[p] == k) ? 1.0f : -1.0f;
            float diff  = inner - mapk;
            acc += diff * diff;           // count each pixel once
        }
    }

    // deterministic block reduction
    __shared__ float red[256];
    red[tid] = acc;
    __syncthreads();
#pragma unroll
    for (int s = 128; s > 0; s >>= 1) {
        if (tid < s) red[tid] += red[tid + s];
        __syncthreads();
    }
    if (tid == 0)
        partial_ws[((size_t)k * NBATCH + b) * TILES_PER_PAIR + tile] = red[0];
}

// ---------------- Kernel 3: finalize 5 scalar outputs --------------------------
__global__ void finalize_kernel(const float* __restrict__ partial_ws,
                                const float* __restrict__ err_ws,
                                float* __restrict__ out)
{
    const int t = threadIdx.x;            // 32 threads: one per (k,b)
    __shared__ float sh[32];
    float per = 0.0f;
    if (t < 32) {
        float s = 0.0f;
        for (int i = 0; i < TILES_PER_PAIR; ++i) s += partial_ws[t * TILES_PER_PAIR + i];
        per = (s / (float)NPIX) * (1.0f - err_ws[t]);
    }
    sh[t] = per;
    __syncthreads();
    if (t == 0) {
        float tot = 0.0f;
        for (int kk = 0; kk < NCLS; ++kk) {
            float s = 0.0f;
            for (int bb = 0; bb < NBATCH; ++bb) s += sh[kk * NBATCH + bb];
            float lk = s / (float)NBATCH;
            out[kk] = lk;
            tot += lk;
        }
        out[4] = tot / (float)NCLS;
    }
}

extern "C" void kernel_launch(void* const* d_in, const int* in_sizes, int n_in,
                              void* d_out, int out_size, void* d_ws, size_t ws_size,
                              hipStream_t stream)
{
    (void)in_sizes; (void)n_in; (void)out_size; (void)ws_size;
    // setup_inputs() dict order:
    const float* f1_0 = (const float*)d_in[0];
    const float* f1_1 = (const float*)d_in[1];
    const float* f1_2 = (const float*)d_in[2];
    const float* f1_3 = (const float*)d_in[3];
    const float* f2_0 = (const float*)d_in[4];
    const float* f2_1 = (const float*)d_in[5];
    const float* f2_2 = (const float*)d_in[6];
    const float* f2_3 = (const float*)d_in[7];
    const int*   t1   = (const int*)d_in[8];
    const int*   t2   = (const int*)d_in[9];

    float* ws      = (float*)d_ws;
    float* y_ws    = ws;             // 2048 floats
    float* err_ws  = ws + 2048;      // 32 floats
    float* part_ws = ws + 2080;      // 1024 floats
    float* out     = (float*)d_out;  // 5 floats

    dim3 g1(NBATCH, NCLS, 1);
    sample_y_kernel<<<g1, 256, 0, stream>>>(f1_0, f1_1, f1_2, f1_3, t1, y_ws, err_ws);

    dim3 g2(TILES_PER_PAIR, NBATCH, NCLS);
    inner_loss_kernel<<<g2, 256, 0, stream>>>(f2_0, f2_1, f2_2, f2_3, t2, y_ws, part_ws);

    finalize_kernel<<<1, 32, 0, stream>>>(part_ws, err_ws, out);
}
